// Knowledge_Decomposition_31181462569636
// MI455X (gfx1250) — matrix operand order repaired and checked by reading, not compiled, not run on hardware
//
#include <hip/hip_runtime.h>

typedef __attribute__((ext_vector_type(16))) __bf16 bfvec16;
typedef __attribute__((ext_vector_type(8)))  __bf16 bfvec8;
typedef __attribute__((ext_vector_type(8)))  float  fvec8;

namespace {
constexpr int kD      = 256;
constexpr int kRows   = 2048;           // B*L = 128*16
constexpr int kMat    = kRows * kD;     // 524288 elements
constexpr int kWElems = kD * kD;        // 65536 elements
constexpr int kChunkK = 32;             // K per WMMA
constexpr int kSlabHalves = kD * kChunkK;  // 8192 bf16 = 16 KB per k-slab
}

// ---------------------------------------------------------------------------
// f32 -> bf16 (round to nearest even)
// ---------------------------------------------------------------------------
__global__ __launch_bounds__(256) void cvt_f32_bf16(const float* __restrict__ in,
                                                    unsigned short* __restrict__ out,
                                                    int n) {
  for (int i = blockIdx.x * blockDim.x + threadIdx.x; i < n;
       i += gridDim.x * blockDim.x) {
    unsigned u = __builtin_bit_cast(unsigned, in[i]);
    u += 0x7FFFu + ((u >> 16) & 1u);
    out[i] = (unsigned short)(u >> 16);
  }
}

// ---------------------------------------------------------------------------
// Fused  out = ReLU(LayerNorm(X @ W^T + b) * gamma + beta)
//
// Block = 128 threads (4 waves). Each wave computes a 16-row M tile across the
// full N=256 strip (16 accumulator tiles). The weight k-slab W[:,k0:k0+32]
// (16 KB bf16) is staged global->LDS with async-to-LDS DMA, double buffered,
// and reused by all 4 waves; B fragments are served by ds_load_b128.
// ---------------------------------------------------------------------------
__global__ __launch_bounds__(128) void mlp_block_gemm_ln(
    const unsigned short* __restrict__ Xbf,
    const unsigned short* __restrict__ Wbf,
    const float* __restrict__ bias,
    const float* __restrict__ gamma,
    const float* __restrict__ beta,
    float* __restrict__ out) {
  __shared__ __align__(16) unsigned short wslab[2][kSlabHalves];  // 32 KB

  const int tid  = threadIdx.x;
  const int lane = tid & 31;
  const int wave = tid >> 5;
  const int r    = lane & 15;
  const int hi   = lane >> 4;                 // half-wave select
  const int m0   = blockIdx.x * 64 + wave * 16;

  const __bf16* xrow  = (const __bf16*)Xbf + (size_t)(m0 + r) * kD;
  const __bf16* wbase = (const __bf16*)Wbf;

  // LDS byte offsets of the two slabs (flat LDS addr low 32 bits == offset).
  const unsigned lds_slab0 = (unsigned)(uintptr_t)&wslab[0][0];
  const unsigned lds_slab1 = (unsigned)(uintptr_t)&wslab[1][0];

  // Stage W[:, k0:k0+32] into slab `buf`: row n occupies 64 bytes at n*64.
  // 128 threads x 128 bytes: thread t stages rows 2t and 2t+1 (8 x b128).
  auto stage = [&](int k0, unsigned lds_base) {
    const __bf16* g0 = wbase + (size_t)(tid * 2) * kD + k0;
    const __bf16* g1 = g0 + kD;
    const unsigned l0 = lds_base + (unsigned)tid * 128u;
#pragma unroll
    for (int i = 0; i < 4; ++i) {
      asm volatile("global_load_async_to_lds_b128 %0, %1, off"
                   :: "v"(l0 + i * 16u), "v"(g0 + i * 8) : "memory");
    }
#pragma unroll
    for (int i = 0; i < 4; ++i) {
      asm volatile("global_load_async_to_lds_b128 %0, %1, off"
                   :: "v"(l0 + 64u + i * 16u), "v"(g1 + i * 8) : "memory");
    }
  };

  fvec8 acc[16];
#pragma unroll
  for (int n = 0; n < 16; ++n)
    acc[n] = fvec8{0.f, 0.f, 0.f, 0.f, 0.f, 0.f, 0.f, 0.f};

  // Prologue: stage chunk 0, wait ASYNCcnt, make it visible to all waves.
  stage(0, lds_slab0);
  asm volatile("s_wait_asynccnt 0" ::: "memory");
  __syncthreads();

#pragma unroll
  for (int c = 0; c < 8; ++c) {
    const int k0 = c * kChunkK;

    // Prefetch next slab into the other buffer while computing this one.
    if (c < 7) stage(k0 + kChunkK, (c & 1) ? lds_slab0 : lds_slab1);

    // A fragment: lanes 0-15 hold K = k0+{0..15},
    // lanes 16-31 hold K = k0+{16..31} for row m0+r.
    bfvec16 afrag = *(const bfvec16*)(xrow + k0 + hi * 16);

    const __bf16* slab = (const __bf16*)&wslab[c & 1][0];
#pragma unroll
    for (int n = 0; n < 16; ++n) {
      // B fragment (K x N): lane = column n*16+r, 16 contiguous K per lane,
      // half-wave selects K block of 16. Row n of the slab is at n*32 halves.
      bfvec16 bfrag = *(const bfvec16*)(slab + (n * 16 + r) * kChunkK + hi * 16);
      acc[n] = __builtin_amdgcn_wmma_f32_16x16x32_bf16(
          false, afrag, false, bfrag, (short)0, acc[n], false, false);
    }

    if (c < 7) {
      asm volatile("s_wait_asynccnt 0" ::: "memory");
      __syncthreads();
    }
  }

  // Per-lane column params: col = n*16 + r (same for every accumulator row j).
  float bv[16], gv[16], btv[16];
#pragma unroll
  for (int n = 0; n < 16; ++n) {
    bv[n]  = bias[n * 16 + r];
    gv[n]  = gamma[n * 16 + r];
    btv[n] = beta[n * 16 + r];
  }

  // C layout: acc[n][j] is row (m0 + j + 8*hi), col (n*16 + r).
  // Each row lives in one 16-lane half-wave -> LN reduction via xor shuffles.
#pragma unroll
  for (int j = 0; j < 8; ++j) {
    float vals[16];
    float s = 0.f, s2 = 0.f;
#pragma unroll
    for (int n = 0; n < 16; ++n) {
      float v = acc[n][j] + bv[n];
      vals[n] = v;
      s  += v;
      s2 += v * v;
    }
#pragma unroll
    for (int mask = 1; mask <= 8; mask <<= 1) {
      s  += __shfl_xor(s,  mask, 32);
      s2 += __shfl_xor(s2, mask, 32);
    }
    const float mean = s * (1.f / kD);
    const float var  = s2 * (1.f / kD) - mean * mean;
    const float rstd = rsqrtf(var + 1e-5f);

    float* orow = out + (size_t)(m0 + j + hi * 8) * kD;
#pragma unroll
    for (int n = 0; n < 16; ++n) {
      float v = (vals[n] - mean) * rstd * gv[n] + btv[n];
      orow[n * 16 + r] = fmaxf(v, 0.f);
    }
  }
}

// ---------------------------------------------------------------------------
// Interaction combine:
//   out = p_align * sigmoid(p_align*(g_align.awp)+abp)
//       + g_align * sigmoid(g_align*(p_align.awg)+abg)
// One wave32 per row of 256; 8 columns per lane; full-wave dot reduction.
// ---------------------------------------------------------------------------
__global__ __launch_bounds__(256) void interaction_combine(
    const float* __restrict__ g_align,
    const float* __restrict__ p_align,
    const float* __restrict__ awg, const float* __restrict__ abg,
    const float* __restrict__ awp, const float* __restrict__ abp,
    float* __restrict__ out) {
  const int lane = threadIdx.x & 31;
  const int wave = threadIdx.x >> 5;
  const int row  = blockIdx.x * 8 + wave;
  const int c0   = lane * 8;

  const float* g = g_align + (size_t)row * kD;
  const float* p = p_align + (size_t)row * kD;

  float gvv[8], pvv[8], wgv[8], wpv[8];
#pragma unroll
  for (int i = 0; i < 8; ++i) {
    gvv[i] = g[c0 + i];
    pvv[i] = p[c0 + i];
    wgv[i] = awg[c0 + i];
    wpv[i] = awp[c0 + i];
  }

  float dot_pg = 0.f, dot_gp = 0.f;
#pragma unroll
  for (int i = 0; i < 8; ++i) {
    dot_pg += pvv[i] * wgv[i];   // p_align . awg
    dot_gp += gvv[i] * wpv[i];   // g_align . awp
  }
#pragma unroll
  for (int mask = 1; mask <= 16; mask <<= 1) {
    dot_pg += __shfl_xor(dot_pg, mask, 32);
    dot_gp += __shfl_xor(dot_gp, mask, 32);
  }

  const float bg = abg[0];
  const float bp = abp[0];
  float* orow = out + (size_t)row * kD;
#pragma unroll
  for (int i = 0; i < 8; ++i) {
    float ga = 1.f / (1.f + __expf(-(gvv[i] * dot_pg + bg)));
    float pa = 1.f / (1.f + __expf(-(pvv[i] * dot_gp + bp)));
    orow[c0 + i] = pvv[i] * pa + gvv[i] * ga;
  }
}

// ---------------------------------------------------------------------------
extern "C" void kernel_launch(void* const* d_in, const int* in_sizes, int n_in,
                              void* d_out, int out_size, void* d_ws, size_t ws_size,
                              hipStream_t stream) {
  (void)in_sizes; (void)n_in; (void)out_size; (void)ws_size;

  const float* gfeat = (const float*)d_in[0];
  const float* pfeat = (const float*)d_in[1];

  // Workspace layout (bytes):
  //   [0)              gfeat bf16  (1 MB)
  //   [1 MB)           pfeat bf16  (1 MB)
  //   [2 MB)           6 weight matrices bf16 (6 * 128 KB)
  //   [2,883,584)      4 align matrices f32 (4 * 2 MB)
  char* ws = (char*)d_ws;
  unsigned short* gfeat_bf = (unsigned short*)ws;
  unsigned short* pfeat_bf = gfeat_bf + kMat;
  unsigned short* Wbf[6];
  {
    unsigned short* cur = pfeat_bf + kMat;
    for (int i = 0; i < 6; ++i) { Wbf[i] = cur; cur += kWElems; }
  }
  float* aligns = (float*)(ws + (size_t)2 * kMat * 2 + (size_t)6 * kWElems * 2);
  float* cga = aligns + 0 * (size_t)kMat;   // common g_align  (from pfeat)
  float* cpa = aligns + 1 * (size_t)kMat;   // common p_align  (from gfeat)
  float* sga = aligns + 2 * (size_t)kMat;   // synergy g_align
  float* spa = aligns + 3 * (size_t)kMat;   // synergy p_align

  // Weight input indices: gs_W, ps_W, c_g_W, c_p_W, s_g_W, s_p_W
  const int widx[6] = {2, 6, 10, 14, 22, 26};

  cvt_f32_bf16<<<512, 256, 0, stream>>>(gfeat, gfeat_bf, kMat);
  cvt_f32_bf16<<<512, 256, 0, stream>>>(pfeat, pfeat_bf, kMat);
  for (int i = 0; i < 6; ++i)
    cvt_f32_bf16<<<64, 256, 0, stream>>>((const float*)d_in[widx[i]], Wbf[i], kWElems);

  float* out     = (float*)d_out;
  float* common  = out + 0 * (size_t)kMat;
  float* synergy = out + 1 * (size_t)kMat;
  float* g_spec  = out + 2 * (size_t)kMat;
  float* p_spec  = out + 3 * (size_t)kMat;

  const dim3 gblk(kRows / 64);   // 32 blocks x 4 waves per MLP block

  // g_spec = MLP(gfeat; gs_*) ; p_spec = MLP(pfeat; ps_*)
  mlp_block_gemm_ln<<<gblk, 128, 0, stream>>>(gfeat_bf, Wbf[0],
      (const float*)d_in[3], (const float*)d_in[4], (const float*)d_in[5], g_spec);
  mlp_block_gemm_ln<<<gblk, 128, 0, stream>>>(pfeat_bf, Wbf[1],
      (const float*)d_in[7], (const float*)d_in[8], (const float*)d_in[9], p_spec);

  // interaction(pfeat, gfeat): g_align = MLP(pfeat; *_g_*), p_align = MLP(gfeat; *_p_*)
  mlp_block_gemm_ln<<<gblk, 128, 0, stream>>>(pfeat_bf, Wbf[2],
      (const float*)d_in[11], (const float*)d_in[12], (const float*)d_in[13], cga);
  mlp_block_gemm_ln<<<gblk, 128, 0, stream>>>(gfeat_bf, Wbf[3],
      (const float*)d_in[15], (const float*)d_in[16], (const float*)d_in[17], cpa);
  mlp_block_gemm_ln<<<gblk, 128, 0, stream>>>(pfeat_bf, Wbf[4],
      (const float*)d_in[23], (const float*)d_in[24], (const float*)d_in[25], sga);
  mlp_block_gemm_ln<<<gblk, 128, 0, stream>>>(gfeat_bf, Wbf[5],
      (const float*)d_in[27], (const float*)d_in[28], (const float*)d_in[29], spa);

  interaction_combine<<<kRows / 8, 256, 0, stream>>>(cga, cpa,
      (const float*)d_in[18], (const float*)d_in[19],
      (const float*)d_in[20], (const float*)d_in[21], common);
  interaction_combine<<<kRows / 8, 256, 0, stream>>>(sga, spa,
      (const float*)d_in[30], (const float*)d_in[31],
      (const float*)d_in[32], (const float*)d_in[33], synergy);
}
